// CustomLSTMCell_31688268710611
// MI455X (gfx1250) — matrix operand order repaired and checked by reading, not compiled, not run on hardware
//
#include <hip/hip_runtime.h>
#include <hip/hip_bf16.h>

typedef __attribute__((ext_vector_type(16))) __bf16 v16bf;
typedef __attribute__((ext_vector_type(8)))  float  v8f;

#define BDIM 256
#define MT 64      // rows per block
#define FT 32      // f-columns per block (x4 gates internally)
#define KC 32      // K chunk
#define SA 40      // LDS A row stride (ushorts): 80B rows -> 16B aligned, bank-spread
#define SB 40      // LDS B row stride (ushorts)
#define ASZ (MT * SA)        // one A buffer (ushorts)
#define BSZ (4 * FT * SB)    // one B buffer (ushorts)

constexpr int    DFF = 1024;
constexpr int    DZ  = 512;
constexpr size_t S   = (size_t)256 * 64 * 1024;   // elems per output tensor

// ---- split workgroup barrier: signal after LDS stores drain, wait late ----
#if __has_builtin(__builtin_amdgcn_s_barrier_signal) && __has_builtin(__builtin_amdgcn_s_barrier_wait)
#define BAR_SIGNAL() do { asm volatile("s_wait_dscnt 0x0" ::: "memory"); \
                          __builtin_amdgcn_s_barrier_signal(-1); } while (0)
#define BAR_WAIT()   do { __builtin_amdgcn_s_barrier_wait(-1);           \
                          asm volatile("" ::: "memory"); } while (0)
#else
#define BAR_SIGNAL() do { } while (0)
#define BAR_WAIT()   __syncthreads()
#endif

// pack two f32 -> packed bf16 dword using the native CDNA5 convert
// (clang-22 does not declare the builtin for gfx1250; emit the ISA op directly)
__device__ __forceinline__ unsigned pk2(float a, float b) {
    unsigned r;
    asm("v_cvt_pk_bf16_f32 %0, %1, %2" : "=v"(r) : "v"(a), "v"(b));
    return r;   // low 16 = bf16(a), high 16 = bf16(b)
}

__device__ __forceinline__ float fast_rcp(float x) {
    return __builtin_amdgcn_rcpf(x);
}

union FragU { uint4 q[2]; v16bf v; };

__global__ __launch_bounds__(BDIM)
void xlstm_fused_kernel(
    const float* __restrict__ Zi, const float* __restrict__ Ci,
    const float* __restrict__ Mi, const float* __restrict__ Hi,
    const float* __restrict__ Ni, const float* __restrict__ mG,
    const float* __restrict__ WIw, const float* __restrict__ WIb,
    const float* __restrict__ WFw, const float* __restrict__ WFb,
    const float* __restrict__ WOw, const float* __restrict__ WOb,
    const float* __restrict__ WZw, const float* __restrict__ WZb,
    const float* __restrict__ RIw, const float* __restrict__ RIb,
    const float* __restrict__ RFw, const float* __restrict__ RFb,
    const float* __restrict__ ROw, const float* __restrict__ ROb,
    const float* __restrict__ RZw, const float* __restrict__ RZb,
    float* __restrict__ out)
{
    __shared__ __align__(16) unsigned short As[2 * ASZ];   // 10 KB (double buffered)
    __shared__ __align__(16) unsigned short Bs[2 * BSZ];   // 20 KB (double buffered)

    const int tid    = threadIdx.x;
    const int lane   = tid & 31;
    const int ln     = lane & 15;
    const int laneHi = lane >> 4;
    const int wid    = tid >> 5;
    const int wy     = wid >> 1;   // 0..3 : 16-row sub-tile
    const int wx     = wid & 1;    // 0..1 : 16-col sub-tile

    const int m0 = blockIdx.x * MT;
    const int f0 = blockIdx.y * FT;

    // ---- staging indices ----
    const int ar = tid >> 2;            // A row 0..63
    const int ak = (tid & 3) * 8;       // 0,8,16,24
    const int fgrp  = tid & 15;         // 16 f-groups of 8 cols = 128 cols
    const int bk2   = (tid >> 4) * 2;   // 0,2,..,30 (two consecutive k rows)
    const int fcol0 = fgrp * 8;         // 0..120
    const int bg    = fgrp >> 2;        // gate 0..3
    const int bfin  = (fgrp & 3) * 8;   // 0,8,16,24 within gate's 32 cols

    // branchless per-thread gate weight selection
    const float* selH = (bg == 0) ? WIw : (bg == 1) ? WFw : (bg == 2) ? WOw : WZw;
    const float* selR = (bg == 0) ? RIw : (bg == 1) ? RFw : (bg == 2) ? ROw : RZw;

    // running source pointers
    const float* paH = Hi + (size_t)(m0 + ar) * DFF + ak;
    const float* paZ = Zi + (size_t)(m0 + ar) * DZ  + ak;
    const float* pbH = selH + (size_t)bk2 * DFF + f0 + bfin;
    const float* pbR = selR + (size_t)bk2 * DFF + f0 + bfin;

    v8f accI = {}, accF = {}, accO = {}, accZ = {};

    float aReg[8];
    float bReg0[8], bReg1[8];

    auto fetch = [&](const float* pa, const float* pb) {
        float4 a0 = *(const float4*)pa;
        float4 a1 = *(const float4*)(pa + 4);
        aReg[0]=a0.x; aReg[1]=a0.y; aReg[2]=a0.z; aReg[3]=a0.w;
        aReg[4]=a1.x; aReg[5]=a1.y; aReg[6]=a1.z; aReg[7]=a1.w;
        float4 b0a = *(const float4*)pb;
        float4 b0b = *(const float4*)(pb + 4);
        float4 b1a = *(const float4*)(pb + DFF);
        float4 b1b = *(const float4*)(pb + DFF + 4);
        bReg0[0]=b0a.x; bReg0[1]=b0a.y; bReg0[2]=b0a.z; bReg0[3]=b0a.w;
        bReg0[4]=b0b.x; bReg0[5]=b0b.y; bReg0[6]=b0b.z; bReg0[7]=b0b.w;
        bReg1[0]=b1a.x; bReg1[1]=b1a.y; bReg1[2]=b1a.z; bReg1[3]=b1a.w;
        bReg1[4]=b1b.x; bReg1[5]=b1b.y; bReg1[6]=b1b.z; bReg1[7]=b1b.w;
    };

    auto commit = [&](int buf) {   // convert staged regs -> bf16 LDS buffer
        uint4 w;
        w.x = pk2(aReg[0], aReg[1]);
        w.y = pk2(aReg[2], aReg[3]);
        w.z = pk2(aReg[4], aReg[5]);
        w.w = pk2(aReg[6], aReg[7]);
        *(uint4*)&As[buf * ASZ + ar * SA + ak] = w;                   // ds_store_b128
        #pragma unroll
        for (int i = 0; i < 8; ++i) {                                 // Bs[f][k2,k2+1]
            unsigned u = pk2(bReg0[i], bReg1[i]);
            *(unsigned*)&Bs[buf * BSZ + (fcol0 + i) * SB + bk2] = u;  // ds_store_b32 (2addr-paired)
        }
    };

    const float* pa = paH;
    const float* pb = pbH;
    fetch(pa, pb);
    commit(0);
    BAR_SIGNAL();                       // round 0: my stores are in

    for (int c = 0; c < 48; ++c) {
        if (c < 47) {                   // issue next chunk's global loads (no wait yet)
            if (c == 31) { pa = paZ; pb = pbR; }          // H-phase -> Z-phase
            else         { pa += KC; pb += (size_t)KC * DFF; }
            fetch(pa, pb);
        }
        BAR_WAIT();                     // buf[c&1] ready (loads still in flight)

        const int buf = c & 1;
        // A fragment: lane(0-15)=row, K in {0..15} (+16 for lanes 16-31)
        FragU af;
        {
            const unsigned short* p = &As[buf * ASZ + (wy * 16 + ln) * SA + laneHi * 16];
            af.q[0] = *(const uint4*)p;
            af.q[1] = *(const uint4*)(p + 8);
        }
        // all four gates' B fragments in distinct registers
        FragU bf0, bf1, bf2, bf3;
        {
            const unsigned short* pB = &Bs[buf * BSZ + (wx * 16 + ln) * SB + laneHi * 16];
            bf0.q[0] = *(const uint4*)(pB);
            bf0.q[1] = *(const uint4*)(pB + 8);
            bf1.q[0] = *(const uint4*)(pB + 1 * FT * SB);
            bf1.q[1] = *(const uint4*)(pB + 1 * FT * SB + 8);
            bf2.q[0] = *(const uint4*)(pB + 2 * FT * SB);
            bf2.q[1] = *(const uint4*)(pB + 2 * FT * SB + 8);
            bf3.q[0] = *(const uint4*)(pB + 3 * FT * SB);
            bf3.q[1] = *(const uint4*)(pB + 3 * FT * SB + 8);
        }
        accI = __builtin_amdgcn_wmma_f32_16x16x32_bf16(false, af.v, false, bf0.v, (short)0, accI, false, false);
        accF = __builtin_amdgcn_wmma_f32_16x16x32_bf16(false, af.v, false, bf1.v, (short)0, accF, false, false);
        accO = __builtin_amdgcn_wmma_f32_16x16x32_bf16(false, af.v, false, bf2.v, (short)0, accO, false, false);
        accZ = __builtin_amdgcn_wmma_f32_16x16x32_bf16(false, af.v, false, bf3.v, (short)0, accZ, false, false);

        if (c < 47) {
            commit(1 - buf);            // loadcnt wait lands here, after the WMMAs
            BAR_SIGNAL();               // round c+1
        }
    }

    // ---- fused epilogue: lane holds rows (r + 8*laneHi), col (lane&15) ----
    const int f  = f0 + wx * 16 + ln;
    const float bI = WIb[f] + RIb[f];
    const float bF = WFb[f] + RFb[f];
    const float bO = WOb[f] + ROb[f];
    const float bZ = WZb[f] + RZb[f];

    #pragma unroll
    for (int r = 0; r < 8; ++r) {
        const int    mrow = m0 + wy * 16 + laneHi * 8 + r;
        const size_t idx  = (size_t)mrow * DFF + f;

        const float mi = Mi[idx];
        const float ni = Ni[idx];
        const float ci = Ci[idx];
        const float hi = Hi[idx];
        const float mv = mG[mrow];

        const float It = accI[r] + bI;
        const float Ft = accF[r] + bF;
        const float Ot = accO[r] + bO;
        const float Zt = accZ[r] + bZ;

        const float Mt  = fmaxf(Ft + mi, It);
        const float Ig  = __expf(It - Mt);
        const float Fg  = __expf(Ft + mi - Mt);
        const float Og  = fast_rcp(1.0f + __expf(-Ot));       // sigmoid
        const float Ntv = Fg * ni + Ig;

        // fast tanh: sign-safe, exp-based
        const float az  = fabsf(Zt);
        const float e2  = __expf(-2.0f * az);
        const float Zg  = copysignf((1.0f - e2) * fast_rcp(1.0f + e2), Zt);

        const float Ctv = (ci * Fg + Zg * Ig) * mv + (1.0f - mv) * ci;
        const float Htv = Og * (Ctv * fast_rcp(Ntv)) * mv + (1.0f - mv) * hi;

        out[idx]         = Ctv;   // C_t
        out[S + idx]     = Mt;    // M_t
        out[2 * S + idx] = Htv;   // H_t
        out[3 * S + idx] = Ntv;   // N_t
    }
}

extern "C" void kernel_launch(void* const* d_in, const int* in_sizes, int n_in,
                              void* d_out, int out_size, void* d_ws, size_t ws_size,
                              hipStream_t stream) {
    const float* p[22];
    for (int i = 0; i < 22; ++i) p[i] = (const float*)d_in[i];
    dim3 grid(16384 / MT, 1024 / FT);   // 256 x 32 blocks
    xlstm_fused_kernel<<<grid, BDIM, 0, stream>>>(
        p[0], p[1], p[2], p[3], p[4], p[5],
        p[6], p[7], p[8], p[9], p[10], p[11], p[12], p[13],
        p[14], p[15], p[16], p[17], p[18], p[19], p[20], p[21],
        (float*)d_out);
}
